// NCDEClassifier_386547057369
// MI455X (gfx1250) — matrix operand order repaired and checked by reading, not compiled, not run on hardware
//
#include <hip/hip_runtime.h>
#include <hip/hip_bf16.h>
#include <math.h>

// Problem constants (from reference)
#define BATCH 512
#define TSTEPS 2048
#define DDIM 32
#define MDIM 64
#define WDIM 128

// LDS pitches (padded so fragment reads spread across all 64 banks)
#define HP 136   // pitch for 128-wide bf16 activation rows
#define YP 72    // pitch for 64-wide bf16 state rows
#define XP 34    // pitch for 32-wide fp32 dXdt rows
#define W2LDS_ROWS 1024
#define W2LDS_BYTES (W2LDS_ROWS * HP * 2)   // 278,528 B dynamic LDS (<320KB/WG)

#define NTHREADS 256                         // 8 waves -> ~512 VGPR budget/wave

typedef __attribute__((ext_vector_type(16))) __bf16          v16bf;
typedef __attribute__((ext_vector_type(8)))  float           v8f;
typedef __attribute__((ext_vector_type(8)))  unsigned short  u16x8;
typedef __attribute__((ext_vector_type(16))) unsigned short  u16x16;

__device__ __forceinline__ unsigned short f2bf(float f) {
    unsigned int u = __float_as_uint(f);
    u += 0x7FFFu + ((u >> 16) & 1u);          // round-to-nearest-even
    return (unsigned short)(u >> 16);
}

// Branchless transcendentals on the hardware TRANS pipe (v_exp_f32 = 2^x).
__device__ __forceinline__ float exp2_hw(float x) { return __builtin_amdgcn_exp2f(x); }
__device__ __forceinline__ float rcp_hw(float x)  { return __builtin_amdgcn_rcpf(x); }

__device__ __forceinline__ float tanh_fast(float x) {
    // tanh(x) = 1 - 2/(e^{2x}+1); exp2 overflow->inf gives exact +/-1 saturation
    float e = exp2_hw(2.8853900817779268f * x);      // e^{2x}
    return 1.0f - 2.0f * rcp_hw(e + 1.0f);
}
__device__ __forceinline__ float gelu_fast(float x) {
    // tanh-gelu == x * sigmoid(1.595769122*(x+0.044715x^3))
    float u = x + 0.044715f * x * x * x;
    return x * rcp_hw(1.0f + exp2_hw(-2.30220824f * u));
}
__device__ __forceinline__ float sigmoid_fast(float x) {
    return rcp_hw(1.0f + exp2_hw(-1.4426950408889634f * x));
}

__device__ __forceinline__ v16bf make_frag(u16x8 lo, u16x8 hi) {
    u16x16 t = __builtin_shufflevector(lo, hi, 0, 1, 2, 3, 4, 5, 6, 7,
                                       8, 9, 10, 11, 12, 13, 14, 15);
    return __builtin_bit_cast(v16bf, t);
}

// A-matrix fragment (16x32 bf16) from a bf16 LDS row.
// Lane<16: K = k0+0..15 ; Lane>=16: K = k0+16..31.
__device__ __forceinline__ v16bf ldsA_frag(const unsigned short* row, int k0, int ab) {
    u16x8 lo = *(const u16x8*)(row + k0 + ab);
    u16x8 hi = *(const u16x8*)(row + k0 + 8 + ab);
    return make_frag(lo, hi);
}

// B-matrix fragment (32x16 bf16), B = W^T, W row-major (out x in), from global.
__device__ __forceinline__ v16bf glbB_frag(const unsigned short* Wb, int ldk, int col,
                                           int k0, int cb) {
    const unsigned short* p = Wb + (size_t)col * ldk + k0 + cb;
    u16x8 lo = *(const u16x8*)p;
    u16x8 hi = *(const u16x8*)(p + 8);
    return make_frag(lo, hi);
}

// Same but from the padded LDS copy of W2 (rows 0..1023, pitch HP).
__device__ __forceinline__ v16bf ldsB_frag(const unsigned short* w2l, int col,
                                           int k0, int cb) {
    const unsigned short* p = w2l + col * HP + k0 + cb;
    u16x8 lo = *(const u16x8*)p;
    u16x8 hi = *(const u16x8*)(p + 8);
    return make_frag(lo, hi);
}

__device__ __forceinline__ v8f wmma_bf16(v16bf a, v16bf b, v8f c) {
    return __builtin_amdgcn_wmma_f32_16x16x32_bf16(false, a, false, b, (short)0, c,
                                                   false, false);
}

// One m-value of GEMM3: two 16-col tiles (d 0..15 / 16..31), tanh, einsum with
// dX/dt, 16-lane butterfly reduce, store k[b][m].
__device__ __forceinline__ void gemm3_pair(const v16bf aF[4], const v16bf bt0[4],
                                           const v16bf bt1[4], float bias0, float bias1,
                                           const float (*dXp)[XP], float (*kb)[MDIM],
                                           int m, int l16, int rbase) {
    float sred[8];
#pragma unroll
    for (int r = 0; r < 8; r++) sred[r] = 0.0f;
    {
        v8f acc = {0.f, 0.f, 0.f, 0.f, 0.f, 0.f, 0.f, 0.f};
#pragma unroll
        for (int k = 0; k < 4; k++) acc = wmma_bf16(aF[k], bt0[k], acc);
#pragma unroll
        for (int r = 0; r < 8; r++)
            sred[r] += tanh_fast(acc[r] + bias0) * dXp[rbase + r][l16];
    }
    {
        v8f acc = {0.f, 0.f, 0.f, 0.f, 0.f, 0.f, 0.f, 0.f};
#pragma unroll
        for (int k = 0; k < 4; k++) acc = wmma_bf16(aF[k], bt1[k], acc);
#pragma unroll
        for (int r = 0; r < 8; r++)
            sred[r] += tanh_fast(acc[r] + bias1) * dXp[rbase + r][16 + l16];
    }
#pragma unroll
    for (int r = 0; r < 8; r++) {
        float sv = sred[r];
        sv += __shfl_xor(sv, 1, 32);
        sv += __shfl_xor(sv, 2, 32);
        sv += __shfl_xor(sv, 4, 32);
        sv += __shfl_xor(sv, 8, 32);
        sred[r] = sv;
    }
    if (l16 == 0) {
#pragma unroll
        for (int r = 0; r < 8; r++) kb[rbase + r][m] = sred[r];
    }
}

// ---------------------------------------------------------------------------
__global__ void cvt_bf16_kernel(const float* __restrict__ src,
                                unsigned short* __restrict__ dst, int n) {
    int i = blockIdx.x * blockDim.x + threadIdx.x;
    if (i < n) dst[i] = f2bf(src[i]);
}

// ---------------------------------------------------------------------------
// encoder: xs[:,0] (B,32) -> y0 (B,64), fp32, one block per batch row
__global__ void encoder_kernel(const float* __restrict__ xs,
                               const float* __restrict__ eW0, const float* __restrict__ eb0,
                               const float* __restrict__ eW1, const float* __restrict__ eb1,
                               const float* __restrict__ eW2, const float* __restrict__ eb2,
                               float* __restrict__ y0ws) {
    __shared__ float xb[DDIM];
    __shared__ float h0[WDIM];
    __shared__ float h1[WDIM];
    const int b = blockIdx.x, tid = threadIdx.x;
    if (tid < DDIM) xb[tid] = xs[(size_t)b * TSTEPS * DDIM + tid];
    __syncthreads();
    {
        float a = eb0[tid];
#pragma unroll 8
        for (int j = 0; j < DDIM; j++) a += eW0[tid * DDIM + j] * xb[j];
        h0[tid] = fmaxf(a, 0.0f);
    }
    __syncthreads();
    {
        float a = eb1[tid];
#pragma unroll 8
        for (int j = 0; j < WDIM; j++) a += eW1[tid * WDIM + j] * h0[j];
        h1[tid] = fmaxf(a, 0.0f);
    }
    __syncthreads();
    if (tid < MDIM) {
        float a = eb2[tid];
#pragma unroll 8
        for (int j = 0; j < WDIM; j++) a += eW2[tid * WDIM + j] * h1[j];
        y0ws[(size_t)b * MDIM + tid] = a;
    }
}

// ---------------------------------------------------------------------------
// Persistent NCDE scan: 32 blocks x 256 threads (8 waves); block owns 16
// batch rows. W2 lower half cached in 272KB dynamic LDS, upper half streamed
// from L2 via a 2-deep pair pipeline (one pair of tiles always in flight).
__global__ __launch_bounds__(NTHREADS) void ncde_scan_kernel(
    const float* __restrict__ times, const float* __restrict__ xs,
    const unsigned short* __restrict__ W0bf, const unsigned short* __restrict__ W1bf,
    const unsigned short* __restrict__ W2bf,
    const float* __restrict__ vb0, const float* __restrict__ vb1,
    const float* __restrict__ vb2,
    const float* __restrict__ y0ws,
    const float* __restrict__ decW, const float* __restrict__ decb,
    float* __restrict__ out) {
    extern __shared__ unsigned short w2lds[];        // [1024][HP] padded W2 rows
    __shared__ unsigned short h1A[16][HP];           // bf16 activations (A mats)
    __shared__ unsigned short h2A[16][HP];
    __shared__ unsigned short ystA[16][YP];          // bf16 stage-input state
    __shared__ float ycur[16][MDIM];                 // fp32 integrated state
    __shared__ float kbuf[16][MDIM];                 // k = f(y) @ dX/dt
    __shared__ float dXs[16][XP];

    const int tid  = threadIdx.x;
    const int wave = tid >> 5;                       // 0..7
    const int lane = tid & 31;
    const int l16  = lane & 15;
    const int rbase = (lane < 16) ? 0 : 8;   // C-matrix row base per half-wave
    const int cb    = (lane < 16) ? 0 : 16;  // B-frag K chunk base
    const int ab    = (lane < 16) ? 0 : 16;  // A-frag K base
    const int b0row = blockIdx.x * 16;

    // ---- loop-invariant small-weight fragments (every wave does G1+G2) ----
    v16bf wB0f[2], wB1f[4];
    const int ncol = wave * 16 + l16;                // 0..127
#pragma unroll
    for (int k = 0; k < 2; k++) wB0f[k] = glbB_frag(W0bf, MDIM, ncol, k * 32, cb);
#pragma unroll
    for (int k = 0; k < 4; k++) wB1f[k] = glbB_frag(W1bf, WDIM, ncol, k * 32, cb);
    const float biasn0 = vb0[ncol];
    const float biasn1 = vb1[ncol];
    // biases for this wave's 16 GEMM3 tiles (8 LDS-resident + 8 streamed)
    float biasL[8], biasG[8];
#pragma unroll
    for (int t8 = 0; t8 < 8; t8++) {
        biasL[t8] = vb2[wave * 128 + t8 * 16 + l16];
        biasG[t8] = vb2[1024 + wave * 128 + t8 * 16 + l16];
    }

    // ---- fill LDS copy of W2 rows 0..1023 (padded pitch) ----
    for (int i = tid; i < W2LDS_ROWS * 16; i += NTHREADS) {
        const int row = i >> 4, c = (i & 15) << 3;
        *(u16x8*)(w2lds + row * HP + c) = *(const u16x8*)(W2bf + row * WDIM + c);
    }

    // ---- init y state (4 elements per thread) ----
    {
        const float* src = y0ws + (size_t)blockIdx.x * 16 * MDIM;
        float* yf = &ycur[0][0];
#pragma unroll
        for (int j = 0; j < 4; j++) yf[tid + NTHREADS * j] = src[tid + NTHREADS * j];
    }
    __syncthreads();

    float k1[4], k2[4], k3[4];
#pragma unroll
    for (int j = 0; j < 4; j++) k1[j] = k2[j] = k3[j] = 0.0f;

#pragma unroll 1
    for (int t = 0; t < TSTEPS - 1; t++) {
        const float dt = times[t + 1] - times[t];
        {   // dXdt for this interval; stage input = current state (as bf16)
#pragma unroll
            for (int j = 0; j < 2; j++) {
                const int e = tid + NTHREADS * j;
                const int bb = e >> 5, dd = e & 31;
                const size_t base =
                    (size_t)(b0row + bb) * TSTEPS * DDIM + (size_t)t * DDIM + dd;
                dXs[bb][dd] = (xs[base + DDIM] - xs[base]) / dt;
                if (t + 2 < TSTEPS)                   // prefetch next interval
                    __builtin_prefetch(&xs[base + 2 * DDIM], 0, 1);
            }
            const float* yf = &ycur[0][0];
#pragma unroll
            for (int j = 0; j < 4; j++) {
                const int e = tid + NTHREADS * j;
                ystA[e >> 6][e & 63] = f2bf(yf[e]);
            }
        }
        __syncthreads();

#pragma unroll 1
        for (int s = 0; s < 4; s++) {
            // Issue streamed-W2 pair 0 early: latency hides behind GEMM1+GEMM2.
            v16bf sA0[4], sA1[4];
            {
                const int c0 = 1024 + wave * 128 + l16;
#pragma unroll
                for (int k = 0; k < 4; k++) {
                    sA0[k] = glbB_frag(W2bf, WDIM, c0, k * 32, cb);
                    sA1[k] = glbB_frag(W2bf, WDIM, c0 + 16, k * 32, cb);
                }
            }

            // ---- GEMM1: h1 = gelu(y @ W0^T + b0), one 16-col tile per wave --
            {
                v8f acc = {0.f, 0.f, 0.f, 0.f, 0.f, 0.f, 0.f, 0.f};
#pragma unroll
                for (int k = 0; k < 2; k++) {
                    v16bf a = ldsA_frag(&ystA[l16][0], k * 32, ab);
                    acc = wmma_bf16(a, wB0f[k], acc);
                }
#pragma unroll
                for (int r = 0; r < 8; r++)
                    h1A[rbase + r][ncol] = f2bf(gelu_fast(acc[r] + biasn0));
            }
            __syncthreads();

            // ---- GEMM2: h2 = gelu(h1 @ W1^T + b1), one tile per wave ----
            {
                v8f acc = {0.f, 0.f, 0.f, 0.f, 0.f, 0.f, 0.f, 0.f};
#pragma unroll
                for (int k = 0; k < 4; k++) {
                    v16bf a = ldsA_frag(&h1A[l16][0], k * 32, ab);
                    acc = wmma_bf16(a, wB1f[k], acc);
                }
#pragma unroll
                for (int r = 0; r < 8; r++)
                    h2A[rbase + r][ncol] = f2bf(gelu_fast(acc[r] + biasn1));
            }
            __syncthreads();

            // ---- GEMM3: f = tanh(h2 @ W2^T + b2); k = sum_d f * dX/dt ----
            {
                v16bf aF[4];
#pragma unroll
                for (int k = 0; k < 4; k++) aF[k] = ldsA_frag(&h2A[l16][0], k * 32, ab);

#pragma unroll
                for (int p = 0; p < 4; p++) {
                    v16bf nx0[4], nx1[4];
                    if (p < 3) {    // issue next streamed pair; latency hides
                        const int c0 = 1024 + wave * 128 + (p + 1) * 32 + l16;
#pragma unroll
                        for (int k = 0; k < 4; k++) {
                            nx0[k] = glbB_frag(W2bf, WDIM, c0, k * 32, cb);
                            nx1[k] = glbB_frag(W2bf, WDIM, c0 + 16, k * 32, cb);
                        }
                    }
                    // streamed pair p: m = 32 + wave*4 + p
                    gemm3_pair(aF, sA0, sA1, biasG[2 * p], biasG[2 * p + 1], dXs,
                               kbuf, 32 + wave * 4 + p, l16, rbase);
                    // LDS-resident pair p: m = wave*4 + p (overlaps nx latency)
                    {
                        v16bf bt0[4], bt1[4];
                        const int c0 = wave * 128 + p * 32 + l16;
#pragma unroll
                        for (int k = 0; k < 4; k++) {
                            bt0[k] = ldsB_frag(w2lds, c0, k * 32, cb);
                            bt1[k] = ldsB_frag(w2lds, c0 + 16, k * 32, cb);
                        }
                        gemm3_pair(aF, bt0, bt1, biasL[2 * p], biasL[2 * p + 1], dXs,
                                   kbuf, wave * 4 + p, l16, rbase);
                    }
                    if (p < 3) {
#pragma unroll
                        for (int k = 0; k < 4; k++) { sA0[k] = nx0[k]; sA1[k] = nx1[k]; }
                    }
                }
            }
            __syncthreads();

            // ---- RK4 combination (256 threads x 4 state elements) ----
            {
                const float* kf = &kbuf[0][0];
                float* yf = &ycur[0][0];
#pragma unroll
                for (int j = 0; j < 4; j++) {
                    const int e = tid + NTHREADS * j;
                    const float kv = kf[e];
                    float sv;
                    if (s == 0)      { k1[j] = kv; sv = yf[e] + 0.5f * dt * kv; }
                    else if (s == 1) { k2[j] = kv; sv = yf[e] + 0.5f * dt * kv; }
                    else if (s == 2) { k3[j] = kv; sv = yf[e] + dt * kv; }
                    else {
                        sv = yf[e] + (dt * (1.0f / 6.0f)) *
                                         (k1[j] + 2.f * k2[j] + 2.f * k3[j] + kv);
                        yf[e] = sv;
                    }
                    if (s < 3) ystA[e >> 6][e & 63] = f2bf(sv);
                }
            }
            __syncthreads();
        }
    }

    // ---- decoder + sigmoid ----
    if (tid < 16) {
        float acc = decb[0];
#pragma unroll
        for (int m = 0; m < MDIM; m++) acc += ycur[tid][m] * decW[m];
        out[b0row + tid] = sigmoid_fast(acc);
    }
}

// ---------------------------------------------------------------------------
extern "C" void kernel_launch(void* const* d_in, const int* in_sizes, int n_in,
                              void* d_out, int out_size, void* d_ws, size_t ws_size,
                              hipStream_t stream) {
    const float* times = (const float*)d_in[0];
    const float* xs    = (const float*)d_in[1];
    const float* eW0 = (const float*)d_in[2];  const float* eb0 = (const float*)d_in[3];
    const float* eW1 = (const float*)d_in[4];  const float* eb1 = (const float*)d_in[5];
    const float* eW2 = (const float*)d_in[6];  const float* eb2 = (const float*)d_in[7];
    const float* vW0 = (const float*)d_in[8];  const float* vb0 = (const float*)d_in[9];
    const float* vW1 = (const float*)d_in[10]; const float* vb1 = (const float*)d_in[11];
    const float* vW2 = (const float*)d_in[12]; const float* vb2 = (const float*)d_in[13];
    const float* dW  = (const float*)d_in[14]; const float* db  = (const float*)d_in[15];

    char* ws = (char*)d_ws;
    unsigned short* W0bf = (unsigned short*)(ws);                    // 128*64
    unsigned short* W1bf = (unsigned short*)(ws + 16384);            // 128*128
    unsigned short* W2bf = (unsigned short*)(ws + 16384 + 32768);    // 2048*128
    float*          y0   = (float*)(ws + 16384 + 32768 + 524288);    // 512*64 f32

    cvt_bf16_kernel<<<(128 * 64 + 255) / 256, 256, 0, stream>>>(vW0, W0bf, 128 * 64);
    cvt_bf16_kernel<<<(128 * 128 + 255) / 256, 256, 0, stream>>>(vW1, W1bf, 128 * 128);
    cvt_bf16_kernel<<<(2048 * 128 + 255) / 256, 256, 0, stream>>>(vW2, W2bf, 2048 * 128);
    encoder_kernel<<<BATCH, WDIM, 0, stream>>>(xs, eW0, eb0, eW1, eb1, eW2, eb2, y0);
    ncde_scan_kernel<<<BATCH / 16, NTHREADS, W2LDS_BYTES, stream>>>(
        times, xs, W0bf, W1bf, W2bf, vb0, vb1, vb2, y0, dW, db, (float*)d_out);
}
